// NCP_18511309045952
// MI455X (gfx1250) — matrix operand order repaired and checked by reading, not compiled, not run on hardware
//
#include <hip/hip_runtime.h>

// ---------------------------------------------------------------------------
// NCP / CfC sequential scan on gfx1250 (MI455X).
// 1 workgroup, 3 waves (wave32). z lives in A-fragments (all rows = z, built
// from LDS), weights live in VGPRs as B-fragments (column n = neuron n), so
// each neuron's gate pre-activations land in a distinct lane of D[0] and the
// tanh/sigmoid combine is a single lane-parallel chain per layer.
// 17 v_wmma_f32_16x16x32_f16 per timestep, 2 barriers per timestep.
// ---------------------------------------------------------------------------

typedef __attribute__((ext_vector_type(16))) _Float16 v16h;
typedef __attribute__((ext_vector_type(8)))  float    v8f;

#define NCP_T   65536
#define WAVES   3
#define BLOCK   (WAVES * 32)

struct NcpPtrs { const float* p[28]; };
// p[0]         : x [65536]
// p[1+8L+2g]   : layer L gate g weights [hid, cat]  (g: ff1, ff2, ta, tb)
// p[2+8L+2g]   : layer L gate g bias    [hid]
// p[25+L]      : layer L mask           [hid, cat]
// L0: hid=38 cat=39 | L1: hid=25 cat=63 | L2: hid=1 cat=26

__device__ __forceinline__ float ncp_rcp(float x) { return __builtin_amdgcn_rcpf(x); }
__device__ __forceinline__ float ncp_tanh(float x) {
  float e = __expf(2.0f * x);                  // inf-safe: rcp(inf)=0 -> 1
  return 1.0f - 2.0f * ncp_rcp(e + 1.0f);
}
__device__ __forceinline__ float ncp_sigmoid(float x) {
  return ncp_rcp(1.0f + __expf(-x));
}

// B-fragment (weights, 32x16 K x N): f16 slot q <-> K = kbase + 8*half + q (q<8) | kbase + 16 + 8*half + (q-8),
// N = colbase + mrow.  Column K==cat carries the bias (z has 1.0 there).
// l0perm: layer0 column permutation K<38 -> W col K+1, K==38 -> bias,
// K==39 -> W col 0 (the x input column).
__device__ __forceinline__ v16h ncp_build_w(const float* w, const float* msk,
                                            const float* bias, int hid, int cat,
                                            int colbase, int kbase,
                                            int mrow, int half, bool l0perm) {
  v16h b;
  const int n = colbase + mrow;
#pragma unroll
  for (int q = 0; q < 16; ++q) {
    const int k = kbase + ((q < 8) ? (8 * half + q) : (16 + 8 * half + (q - 8)));
    float v = 0.0f;
    if (n < hid) {
      if (l0perm) {
        if (k < 38)        { v = w[n * cat + (k + 1)]; if (msk) v *= msk[n * cat + (k + 1)]; }
        else if (k == 38)  { v = bias[n]; }
        else if (k == 39)  { v = w[n * cat + 0];       if (msk) v *= msk[n * cat + 0]; }
      } else {
        if (k < cat)       { v = w[n * cat + k];       if (msk) v *= msk[n * cat + k]; }
        else if (k == cat) { v = bias[n]; }
      }
    }
    b[q] = (_Float16)v;
  }
  return b;
}

// A-fragment (z, 16x32): every lane loads the same z slice, so all 16 rows
// equal z and every element of D is a valid neuron output.
// half0: slot q -> K = q (q<8) | q+8 ; half1: K = q+8 | q+16
// -> two aligned 16B LDS loads at kbase + 8*half and kbase + 16 + 8*half.
__device__ __forceinline__ v16h ncp_load_a(const _Float16* z, int kbase, int half) {
  v16h a;
  const _Float16* lo = z + kbase + 8 * half;
  const _Float16* hi = z + kbase + 16 + 8 * half;
#pragma unroll
  for (int q = 0; q < 8; ++q) { a[q] = lo[q]; a[8 + q] = hi[q]; }
  return a;
}

__device__ __forceinline__ v8f ncp_wmma2(v16h a0, v16h b0, v16h a1, v16h b1) {
  v8f c = {};
  c = __builtin_amdgcn_wmma_f32_16x16x32_f16(false, a0, false, b0, (short)0, c, false, false);
  c = __builtin_amdgcn_wmma_f32_16x16x32_f16(false, a1, false, b1, (short)0, c, false, false);
  return c;
}

__global__ __launch_bounds__(BLOCK, 1)
void ncp_cfc_kernel(NcpPtrs P, float* __restrict__ out) {
  const int tid  = threadIdx.x;
  const int wave = tid >> 5;
  const int lane = tid & 31;
  const int half = lane >> 4;
  const int mrow = lane & 15;

  const float* __restrict__ xg = P.p[0];

  // zb1: [h0_new(0..37), h1_state(38..62), 1.0@63], parity double-buffered.
  // zb2: [h1_new(0..24), (h2 kept in register)@25, 1.0@26, 0-pad..31]
  __shared__ __align__(32) _Float16 zb1[2][64];
  __shared__ __align__(32) _Float16 zb2[32];

  for (int i = tid; i < 2 * 64; i += BLOCK) zb1[i >> 6][i & 63] = (_Float16)0.0f;
  for (int i = tid; i < 32; i += BLOCK)     zb2[i] = (_Float16)0.0f;
  if (tid == 0) {
    zb1[0][63] = (_Float16)1.0f;
    zb1[1][63] = (_Float16)1.0f;
    zb2[26]    = (_Float16)1.0f;
  }

  // One-time cache warm so the per-lane fragment-build loads hit L0/L2.
  {
    const int sz[28] = {NCP_T,
      1482, 38, 1482, 38, 1482, 38, 1482, 38,
      1575, 25, 1575, 25, 1575, 25, 1575, 25,
      26, 1, 26, 1, 26, 1, 26, 1,
      1482, 1575, 26};
    for (int a = 1; a < 28; ++a)
      for (int i = tid * 32; i < sz[a]; i += BLOCK * 32)
        __builtin_prefetch(P.p[a] + i, 0, 0);
  }

  // ---- Weight B-fragments, register-resident for all 65536 steps ----
  v16h W0[4][2];                 // layer0: neuron tile j = wave (cols 16j..16j+15 of 38)
#pragma unroll
  for (int g = 0; g < 4; ++g) {
    const float* w  = P.p[1 + 2 * g];
    const float* b  = P.p[2 + 2 * g];
    const float* mk = (g < 2) ? P.p[25] : nullptr;
#pragma unroll
    for (int s = 0; s < 2; ++s)
      W0[g][s] = ncp_build_w(w, mk, b, 38, 39, 16 * wave, 32 * s, mrow, half, true);
  }

  v16h W1[4][2] = {};            // layer1: waves 0,1 (cols 16j..16j+15 of 25)
  if (wave < 2) {
#pragma unroll
    for (int g = 0; g < 4; ++g) {
      const float* w  = P.p[9 + 2 * g];
      const float* b  = P.p[10 + 2 * g];
      const float* mk = (g < 2) ? P.p[26] : nullptr;
#pragma unroll
      for (int s = 0; s < 2; ++s)
        W1[g][s] = ncp_build_w(w, mk, b, 25, 63, 16 * wave, 32 * s, mrow, half, false);
    }
  }

  v16h W2 = {};                  // layer2: wave 0, gate g in column g (N = mrow)
  if (wave == 0) {
#pragma unroll
    for (int q = 0; q < 16; ++q) {
      const int k = (q < 8) ? (8 * half + q) : (16 + 8 * half + (q - 8));   // K: 0..24 h1, 25 h2, 26 bias
      float v = 0.0f;
      if (mrow < 4) {
        const float* w = P.p[17 + 2 * mrow];
        const float* b = P.p[18 + 2 * mrow];
        if (k < 26)       { v = w[k]; if (mrow < 2) v *= P.p[27][k]; }
        else if (k == 26) { v = b[0]; }
      }
      W2[q] = (_Float16)v;
    }
  }

  __syncthreads();

  float xcur = xg[0];            // software-pipelined input
  float h2   = 0.0f;             // motor recurrent state (wave0, register)

#pragma clang loop unroll(disable)
  for (int t = 0; t < NCP_T; ++t) {
    if (((t & 31) == 0) && (t + 512 < NCP_T))
      __builtin_prefetch(xg + t + 512, 0, 0);
    const float xnext = xg[(t + 1) & (NCP_T - 1)];

    const int p0 = t & 1, p1 = p0 ^ 1;

    // ------------------- layer 0 : all 3 waves -------------------
    // z0(K): 0..37 = h0 state (zb1[p1][0..37]), 38 = 1.0 (bias), 39 = x_t
    const _Float16* z0 = &zb1[p1][0];
    v16h a00 = ncp_load_a(z0, 0, half);
    v16h a01;
    {
      const _Float16* lo = z0 + 32;            // elems 32..39 (38,39 overwritten)
#pragma unroll
      for (int q = 0; q < 8; ++q) a01[q] = lo[q];
      a01[6] = (_Float16)1.0f;                 // K=38 bias input
      a01[7] = (_Float16)xcur;                 // K=39 x_t
#pragma unroll
      for (int q = 8; q < 16; ++q) a01[q] = (_Float16)0.0f;
      if (half) {                              // half1 covers K=40..47,56..63 -> zero
#pragma unroll
        for (int q = 0; q < 8; ++q) a01[q] = (_Float16)0.0f;
      }
    }

    v8f c0[4];
#pragma unroll
    for (int g = 0; g < 4; ++g)
      c0[g] = ncp_wmma2(a00, W0[g][0], a01, W0[g][1]);

    {
      const float f1 = ncp_tanh(c0[0][0]);
      const float f2 = ncp_tanh(c0[1][0]);
      const float s  = ncp_sigmoid(c0[3][0] - c0[2][0]);
      const float h  = f1 + s * (f2 - f1);
      const int   m  = 16 * wave + mrow;       // halves duplicate: benign dup stores
      if (m < 38) zb1[p0][m] = (_Float16)h;    // h0 state (t+1) == layer1 input (t)
    }
    __syncthreads();                           // zb1[p0][0..37] ready

    // ------------------- layer 1 : waves 0,1 -------------------
    if (wave < 2) {
      const _Float16* z1 = &zb1[p0][0];        // [h0_new, h1_state, 1.0]
      v16h a10 = ncp_load_a(z1, 0, half);
      v16h a11 = ncp_load_a(z1, 32, half);

      v8f c1[4];
#pragma unroll
      for (int g = 0; g < 4; ++g)
        c1[g] = ncp_wmma2(a10, W1[g][0], a11, W1[g][1]);

      const float f1 = ncp_tanh(c1[0][0]);
      const float f2 = ncp_tanh(c1[1][0]);
      const float s  = ncp_sigmoid(c1[3][0] - c1[2][0]);
      const float h  = f1 + s * (f2 - f1);
      const int   m  = 16 * wave + mrow;
      if (m < 25) {
        const _Float16 hv = (_Float16)h;
        zb1[p1][38 + m] = hv;                  // h1 state for step t+1
        zb2[m]          = hv;                  // input to layer2 this step
      }
    }
    __syncthreads();                           // zb2[0..24] ready

    // ------------------- layer 2 : wave 0 -------------------
    if (wave == 0) {
      v16h a2 = ncp_load_a(&zb2[0], 0, half);
      if (half) a2[9] = (_Float16)h2;          // K=25: motor recurrent state
      v8f cz = {};
      v8f c2 = __builtin_amdgcn_wmma_f32_16x16x32_f16(false, a2, false, W2,
                                                      (short)0, cz, false, false);
      // lane n of D[0] = gate n of the motor neuron
      const float v   = c2[0];
      const float pf1 = __shfl(v, 0);
      const float pf2 = __shfl(v, 1);
      const float pta = __shfl(v, 2);
      const float ptb = __shfl(v, 3);
      const float f1  = ncp_tanh(pf1);
      const float f2  = ncp_tanh(pf2);
      const float s   = ncp_sigmoid(ptb - pta);
      const float y   = f1 + s * (f2 - f1);
      h2 = y;
      if (lane == 0) out[t] = y;
    }

    xcur = xnext;
  }
}

extern "C" void kernel_launch(void* const* d_in, const int* in_sizes, int n_in,
                              void* d_out, int out_size, void* d_ws, size_t ws_size,
                              hipStream_t stream) {
  (void)in_sizes; (void)out_size; (void)d_ws; (void)ws_size;
  NcpPtrs P;
  for (int i = 0; i < 28; ++i)
    P.p[i] = (const float*)d_in[(i < n_in) ? i : (n_in - 1)];
  ncp_cfc_kernel<<<1, BLOCK, 0, stream>>>(P, (float*)d_out);
}
